// HarmonicOscillator_33337536152364
// MI455X (gfx1250) — matrix operand order repaired and checked by reading, not compiled, not run on hardware
//
#include <hip/hip_runtime.h>
#include <hip/hip_bf16.h>
#include <math.h>

// ---------------------------------------------------------------------------
// Harmonic additive synth + theoretical LPF, CDNA5 (gfx1250, wave32).
//   audio[b,t] = sum_n amp[b,n] * gain(n*f0, fc) * sin(2*pi*n*cumF0[b,t]/SR)
// cumsum(f0*h) == h*cumsum(f0)  -> only an O(B*T) scan of f0 is needed.
// The 64-way amp-weighted harmonic reduction runs on the matrix pipe as
//   D = A(gain*sin, f16) x B(amp broadcast, f16)   via v_wmma_f32_16x16x32_f16
// co-executing with the VALU/TRANS sin+rsq work.
// ---------------------------------------------------------------------------

#define SR            44100.0f
#define NUM_H         64
#define TILE          128            // time samples per block (8 waves x 16)
#define TWO_PI_F      6.28318530717958647692f

typedef __attribute__((ext_vector_type(16))) _Float16 v16h;
typedef __attribute__((ext_vector_type(8)))  float    v8f;

// ---- Kernel A: per-tile sum of f0 (tile = TILE samples) -------------------
__global__ void ho_tile_reduce(const float* __restrict__ f0,
                               float* __restrict__ partials,
                               int T, int numTiles) {
    const int b    = blockIdx.y;
    const int tile = blockIdx.x;
    const int t    = tile * TILE + threadIdx.x;           // blockDim.x == 128
    float v = (t < T) ? f0[(size_t)b * T + t] : 0.0f;
    __shared__ float s[TILE];
    s[threadIdx.x] = v;
    __syncthreads();
    #pragma unroll
    for (int off = TILE / 2; off > 0; off >>= 1) {
        if (threadIdx.x < off) s[threadIdx.x] += s[threadIdx.x + off];
        __syncthreads();
    }
    if (threadIdx.x == 0) partials[(size_t)b * numTiles + tile] = s[0];
}

// ---- Kernel B: exclusive scan of tile sums (tiny: ~690 per batch) ---------
__global__ void ho_scan_partials(const float* __restrict__ partials,
                                 float* __restrict__ offsets,
                                 int B, int numTiles) {
    const int b = threadIdx.x;
    if (b < B) {
        float run = 0.0f;
        for (int i = 0; i < numTiles; ++i) {
            const size_t idx = (size_t)b * numTiles + i;
            float s = partials[idx];
            offsets[idx] = run;           // exclusive prefix of tile sums
            run += s;
        }
    }
}

// ---- Kernel C: in-tile scan + synthesis, WMMA harmonic reduction ----------
__global__ void ho_synth(const float* __restrict__ f0,
                         const float* __restrict__ hamp,
                         const float* __restrict__ fc,
                         const float* __restrict__ tileOffsets,
                         float* __restrict__ out,
                         int T, int numTiles) {
    const int b    = blockIdx.y;
    const int tile = blockIdx.x;
    const int base = tile * TILE;
    const int tid  = threadIdx.x;                         // blockDim.x == 256

    __shared__ float sF0[TILE];
    __shared__ float sCum[TILE];
    __shared__ float sAmp[NUM_H];

    if (tid < TILE) {
        int t = base + tid;
        sF0[tid] = (t < T) ? f0[(size_t)b * T + t] : 0.0f;
    }
    if (tid < NUM_H) sAmp[tid] = hamp[b * NUM_H + tid];
    __syncthreads();

    // Hillis-Steele inclusive scan of the 128 f0 values (7 steps).
    if (tid < TILE) sCum[tid] = sF0[tid];
    __syncthreads();
    #pragma unroll
    for (int off = 1; off < TILE; off <<= 1) {
        float v = 0.0f;
        if (tid < TILE) v = sCum[tid] + ((tid >= off) ? sCum[tid - off] : 0.0f);
        __syncthreads();
        if (tid < TILE) sCum[tid] = v;
        __syncthreads();
    }

    const float tileOff = tileOffsets[(size_t)b * numTiles + tile];
    const float inv_fc  = 1.0f / (fc[b] + 1e-8f);
    const float invQ    = 1.0f / (0.707f + 1e-8f);
    const float invQ2   = invQ * invQ;
    const float invSR   = 1.0f / SR;

    // wave32: each wave owns 16 time samples (WMMA M dim).
    const int wave  = tid >> 5;
    const int lane  = tid & 31;
    const int m     = lane & 15;           // M index (time within wave)
    const int khalf = lane >> 4;           // which K-half this lane supplies
    const int tloc  = wave * 16 + m;       // 8 waves * 16 = 128 = TILE

    const float cum = sCum[tloc] + tileOff;    // inclusive prefix of f0
    const float f0v = sF0[tloc];

    v8f acc = {};                               // f32 16x16 accumulator
    #pragma unroll
    for (int c = 0; c < 2; ++c) {               // two K=32 harmonic chunks
        v16h a, bm;
        #pragma unroll
        for (int e = 0; e < 16; ++e) {
            // A 16x32 f16 layout: lanes0-15 e<8 -> K=e, e>=8 -> K=e+8;
            // lanes16-31 shifted by +8.
            const int   ka   = ((e < 8) ? e : e + 8) + 8 * khalf;
            const int   h    = 32 * c + ka;                 // 0-based harmonic
            const float n    = (float)(h + 1);
            // LPF theoretical magnitude response
            const float r    = (n * f0v) * inv_fc;
            const float r2   = r * r;
            const float d    = 1.0f - r2;
            const float den  = d * d + r2 * invQ2 + 1e-8f;
            const float g    = __frsqrt_rn(den);            // v_rsq_f32
            // phase in revolutions -> fract -> radians in [0, 2*pi)
            float ph = cum * (n * invSR);
            ph -= floorf(ph);
            const float s = __sinf(TWO_PI_F * ph);          // v_sin_f32
            a[e] = (_Float16)(g * s);
            // B Kx16 f16 layout: same K order as A (e<8 -> K=e, e>=8 -> K=e+8; lanes16-31 +8)
            const int kb = ((e < 8) ? e : e + 8) + 8 * khalf;
            bm[e] = (_Float16)sAmp[32 * c + kb];
        }
        // D = A x B + C : amp-weighted harmonic reduction on the matrix pipe
        acc = __builtin_amdgcn_wmma_f32_16x16x32_f16(
            /*neg_a=*/false, a, /*neg_b=*/false, bm,
            /*c_mod=*/(short)0, acc, /*reuse_a=*/false, /*reuse_b=*/false);
    }

    // Every column of D is identical (B replicated across N). C/D layout:
    // VGPR r holds M=r (lanes 0-15) and M=r+8 (lanes 16-31).
    if (lane == 0 || lane == 16) {
        const int mbase = wave * 16 + khalf * 8;
        #pragma unroll
        for (int r = 0; r < 8; ++r) {
            const int t = base + mbase + r;
            if (t < T) out[(size_t)b * T + t] = acc[r];
        }
    }
}

// ---------------------------------------------------------------------------
extern "C" void kernel_launch(void* const* d_in, const int* in_sizes, int n_in,
                              void* d_out, int out_size, void* d_ws, size_t ws_size,
                              hipStream_t stream) {
    const float* f0   = (const float*)d_in[0];   // (B,1,T) f32
    const float* hamp = (const float*)d_in[1];   // (B,64)  f32
    const float* fc   = (const float*)d_in[2];   // (B,1,1) f32
    float*       out  = (float*)d_out;           // (B,T)   f32

    const int B = in_sizes[2];                   // cutoff count == batch
    const int T = in_sizes[0] / B;
    const int numTiles = (T + TILE - 1) / TILE;

    float* partials = (float*)d_ws;                          // B*numTiles
    float* offsets  = partials + (size_t)B * numTiles;       // B*numTiles
    (void)ws_size; (void)n_in; (void)out_size;

    dim3 grid(numTiles, B);
    ho_tile_reduce<<<grid, TILE, 0, stream>>>(f0, partials, T, numTiles);
    ho_scan_partials<<<1, 32, 0, stream>>>(partials, offsets, B, numTiles);
    ho_synth<<<grid, 256, 0, stream>>>(f0, hamp, fc, offsets, out, T, numTiles);
}
